// BubbleformerAttentionBlock_6622839571044
// MI455X (gfx1250) — matrix operand order repaired and checked by reading, not compiled, not run on hardware
//
#include <hip/hip_runtime.h>

// ---------------------------------------------------------------------------
// Bubbleformer attention block for MI455X (gfx1250, wave32, WMMA).
// Heavy work = two 1x1-conv GEMMs -> bf16 v_wmma_f32_16x16x32_bf16, f32 accum.
// Workspace layout (needs ~197 MB):
//   [0)              xn bf16  (32 x 768 x 1024)           50,331,648 B  (reused as attn-out)
//   [50331648)       w_qkv bf16 (2304 x 768)               3,538,944 B
//   [53870592)       w_out bf16 (768 x 768)                1,179,648 B
//   [55050240)       qkv bf16 (32 x 2304 x 1024)         150,994,944 B
// ---------------------------------------------------------------------------

typedef unsigned short u16;
typedef unsigned int   u32;
typedef __attribute__((ext_vector_type(16))) __bf16 v16bf;
typedef __attribute__((ext_vector_type(8)))  float  v8f;

#define EMB_  768
#define HEADS_ 12
#define HD_    64
#define NTOK_  16
#define HWSZ_  1024
#define BATCH_ 2
#define SAMP_  (BATCH_ * NTOK_)   // 32
#define KDIM_  EMB_

__device__ __forceinline__ u16 f2bf(float f) {
  u32 u = __float_as_uint(f);
  u32 r = u + 0x7FFFu + ((u >> 16) & 1u);   // round-to-nearest-even
  return (u16)(r >> 16);
}
__device__ __forceinline__ float bf2f(u16 h) {
  return __uint_as_float(((u32)h) << 16);
}

// ---------------------------------------------------------------------------
__global__ void cvt_f32_bf16(const float* __restrict__ src, u16* __restrict__ dst, int n) {
  int i = blockIdx.x * blockDim.x + threadIdx.x;
  if (i < n) dst[i] = f2bf(src[i]);
}

// ---------------------------------------------------------------------------
// Instance norm over HW=1024 for one (sample, channel) per block; fp32 -> bf16.
__global__ __launch_bounds__(256) void instnorm_f32(const float* __restrict__ x,
                                                    const float* __restrict__ w,
                                                    const float* __restrict__ b,
                                                    u16* __restrict__ y) {
  const int bc = blockIdx.x;          // s*EMB + c
  const int c  = bc % EMB_;
  const float* xp = x + (size_t)bc * HWSZ_;
  u16* yp = y + (size_t)bc * HWSZ_;
  const int t = threadIdx.x;
  float v0 = xp[t], v1 = xp[t + 256], v2 = xp[t + 512], v3 = xp[t + 768];
  __shared__ float ss[256], sq[256];
  ss[t] = v0 + v1 + v2 + v3;
  sq[t] = v0 * v0 + v1 * v1 + v2 * v2 + v3 * v3;
  __syncthreads();
  for (int off = 128; off > 0; off >>= 1) {
    if (t < off) { ss[t] += ss[t + off]; sq[t] += sq[t + off]; }
    __syncthreads();
  }
  float mu  = ss[0] * (1.0f / 1024.0f);
  float var = sq[0] * (1.0f / 1024.0f) - mu * mu;
  float rs  = rsqrtf(var + 1e-5f);
  float sc  = w[c] * rs;
  float sh  = b[c] - mu * sc;
  yp[t]       = f2bf(v0 * sc + sh);
  yp[t + 256] = f2bf(v1 * sc + sh);
  yp[t + 512] = f2bf(v2 * sc + sh);
  yp[t + 768] = f2bf(v3 * sc + sh);
}

// Same, bf16 in-place.
__global__ __launch_bounds__(256) void instnorm_bf16(u16* __restrict__ y,
                                                     const float* __restrict__ w,
                                                     const float* __restrict__ b) {
  const int bc = blockIdx.x;
  const int c  = bc % EMB_;
  u16* yp = y + (size_t)bc * HWSZ_;
  const int t = threadIdx.x;
  float v0 = bf2f(yp[t]), v1 = bf2f(yp[t + 256]), v2 = bf2f(yp[t + 512]), v3 = bf2f(yp[t + 768]);
  __shared__ float ss[256], sq[256];
  ss[t] = v0 + v1 + v2 + v3;
  sq[t] = v0 * v0 + v1 * v1 + v2 * v2 + v3 * v3;
  __syncthreads();
  for (int off = 128; off > 0; off >>= 1) {
    if (t < off) { ss[t] += ss[t + off]; sq[t] += sq[t + off]; }
    __syncthreads();
  }
  float mu  = ss[0] * (1.0f / 1024.0f);
  float var = sq[0] * (1.0f / 1024.0f) - mu * mu;
  float rs  = rsqrtf(var + 1e-5f);
  float sc  = w[c] * rs;
  float sh  = b[c] - mu * sc;
  yp[t]       = f2bf(v0 * sc + sh);
  yp[t + 256] = f2bf(v1 * sc + sh);
  yp[t + 512] = f2bf(v2 * sc + sh);
  yp[t + 768] = f2bf(v3 * sc + sh);
}

// ---------------------------------------------------------------------------
// WMMA GEMM:  Out[s][o][n] = sum_k Wt[o][k] * X[s][k][n] + bias[o]
// Workgroup tile: 128 (o) x 128 (n), 8 waves (4 m-tiles x 2 n-tiles),
// each wave computes 32 (o) x 64 (n) via 2x4 wmma accumulators.
// K step 32 (bf16 16x16x32). LDS tiles stored in exact per-lane fragment order
// (k = 16*half + i for A and B) so each fragment = one contiguous 32B read per lane.
//   A (16x32 bf16): lane = M%16 + 16*half, elem i: v=i/2,p=i%2,
//                   K = 16*half + 2v + p
//   B (32x16 bf16): lane = N%16 + 16*(K/16), elem i: K = 16*(lane/16) + i
template <bool OUT_BF16>
__global__ __launch_bounds__(256) void gemm_bf16_wmma(const u16* __restrict__ Wt,
                                                      const u16* __restrict__ X,
                                                      const float* __restrict__ bias,
                                                      void* __restrict__ Out, int O) {
  const int s  = blockIdx.x;
  const int o0 = blockIdx.y * 128;
  const int n0 = blockIdx.z * 128;

  __shared__ __align__(32) u16 Alds[8 * 512];   // 8 mblks x (32 lanes x 16 elems)  8 KB
  __shared__ __align__(32) u16 Blds[8 * 512];   // 8 nblks x (32 lanes x 16 elems)  8 KB

  const int t    = threadIdx.x;
  const int lane = t & 31;
  const int wave = t >> 5;
  const int wm   = wave & 3;   // o sub-tile (32 rows)
  const int wn   = wave >> 2;  // n sub-tile (64 cols)

  v8f acc[2][4] = {};

  const u16* Xs = X + (size_t)s * KDIM_ * HWSZ_ + n0;

  for (int k0 = 0; k0 < KDIM_; k0 += 32) {
    __syncthreads();
    // ---- stage A (128x32): 1024 quads of 4 bf16 along K, swizzle-write b64.
#pragma unroll
    for (int qq = 0; qq < 4; ++qq) {
      int q  = t + qq * 256;
      int m  = q >> 3;
      int kq = (q & 7) << 2;
      uint2 d = *(const uint2*)(Wt + (size_t)(o0 + m) * KDIM_ + (k0 + kq));
      int half  = kq >> 4;
      int i0    = kq & 15;
      int laneA = (m & 15) | (half << 4);
      *(uint2*)(Alds + (m >> 4) * 512 + laneA * 16 + i0) = d;
    }
    // ---- stage B (32x128): 1024 quads of 4 bf16 along N, scatter b16 writes.
#pragma unroll
    for (int qq = 0; qq < 4; ++qq) {
      int q  = t + qq * 256;
      int kk = q >> 5;             // 0..31
      int nq = (q & 31) << 2;      // 0..124
      uint2 d = *(const uint2*)(Xs + (size_t)(k0 + kk) * HWSZ_ + nq);
      int i     = kk & 15;
      int laneB = (nq & 15) | ((kk >> 4) << 4);
      u16* dst  = Blds + (nq >> 4) * 512 + laneB * 16 + i;
      dst[0]  = (u16)(d.x & 0xFFFFu);
      dst[16] = (u16)(d.x >> 16);
      dst[32] = (u16)(d.y & 0xFFFFu);
      dst[48] = (u16)(d.y >> 16);
    }
    if (k0 + 32 < KDIM_) {  // prefetch next K tile of X (global_prefetch_b8, near)
      __builtin_prefetch(Xs + (size_t)(k0 + 32 + (t >> 5)) * HWSZ_ + ((t & 31) << 2), 0, 3);
    }
    __syncthreads();
    // ---- fragments: one contiguous 32B LDS read per lane each.
    v16bf a0 = *(const v16bf*)(Alds + (wm * 2 + 0) * 512 + lane * 16);
    v16bf a1 = *(const v16bf*)(Alds + (wm * 2 + 1) * 512 + lane * 16);
    v16bf b0 = *(const v16bf*)(Blds + (wn * 4 + 0) * 512 + lane * 16);
    v16bf b1 = *(const v16bf*)(Blds + (wn * 4 + 1) * 512 + lane * 16);
    v16bf b2 = *(const v16bf*)(Blds + (wn * 4 + 2) * 512 + lane * 16);
    v16bf b3 = *(const v16bf*)(Blds + (wn * 4 + 3) * 512 + lane * 16);
    acc[0][0] = __builtin_amdgcn_wmma_f32_16x16x32_bf16(false, a0, false, b0, (short)0, acc[0][0], false, false);
    acc[0][1] = __builtin_amdgcn_wmma_f32_16x16x32_bf16(false, a0, false, b1, (short)0, acc[0][1], false, false);
    acc[0][2] = __builtin_amdgcn_wmma_f32_16x16x32_bf16(false, a0, false, b2, (short)0, acc[0][2], false, false);
    acc[0][3] = __builtin_amdgcn_wmma_f32_16x16x32_bf16(false, a0, false, b3, (short)0, acc[0][3], false, false);
    acc[1][0] = __builtin_amdgcn_wmma_f32_16x16x32_bf16(false, a1, false, b0, (short)0, acc[1][0], false, false);
    acc[1][1] = __builtin_amdgcn_wmma_f32_16x16x32_bf16(false, a1, false, b1, (short)0, acc[1][1], false, false);
    acc[1][2] = __builtin_amdgcn_wmma_f32_16x16x32_bf16(false, a1, false, b2, (short)0, acc[1][2], false, false);
    acc[1][3] = __builtin_amdgcn_wmma_f32_16x16x32_bf16(false, a1, false, b3, (short)0, acc[1][3], false, false);
  }

  // ---- epilogue. C/D layout: elem r -> M = r + 8*(lane/16), N = lane%16.
  const int half = lane >> 4;
#pragma unroll
  for (int mi = 0; mi < 2; ++mi) {
#pragma unroll
    for (int ni = 0; ni < 4; ++ni) {
      v8f c = acc[mi][ni];
      int obase = o0 + wm * 32 + mi * 16 + half * 8;
      int ncol  = n0 + wn * 64 + ni * 16 + (lane & 15);
#pragma unroll
      for (int r = 0; r < 8; ++r) {
        int o = obase + r;
        float val = c[r] + bias[o];
        size_t idx = ((size_t)s * O + o) * HWSZ_ + ncol;
        if (OUT_BF16) ((u16*)Out)[idx] = f2bf(val);
        else          ((float*)Out)[idx] = val;
      }
    }
  }
}

// ---------------------------------------------------------------------------
// Attention per (b, hw, head): qkv bf16 [S][3*EMB][HW] with channel
// o = head*192 + {q:0-63, k:64-127, v:128-191}. Output bf16 [S][EMB][HW],
// channel = head*64 + e.  16 tokens, head_dim 64 -> LDS + VALU.
__global__ __launch_bounds__(256) void attention_kernel(const u16* __restrict__ qkv,
                                                        u16* __restrict__ outb,
                                                        const float* __restrict__ qw,
                                                        const float* __restrict__ qb,
                                                        const float* __restrict__ kw,
                                                        const float* __restrict__ kb,
                                                        const float* __restrict__ scale) {
  const int hw   = blockIdx.x;
  const int head = blockIdx.y;
  const int b    = blockIdx.z;
  __shared__ float Q[16][64], Kt[16][64], V[16][64];
  __shared__ float S_[16][17];
  const int t = threadIdx.x;

#pragma unroll
  for (int j = 0; j < 12; ++j) {
    int idx   = t + j * 256;        // 0..3071
    int which = idx >> 10;          // 0=q 1=k 2=v
    int rem   = idx & 1023;
    int n = rem >> 6, e = rem & 63;
    size_t g = ((size_t)(b * NTOK_ + n) * (3 * EMB_) + head * (3 * HD_) + which * HD_ + e) * HWSZ_ + hw;
    float f = bf2f(qkv[g]);
    if (which == 0) Q[n][e] = f;
    else if (which == 1) Kt[n][e] = f;
    else V[n][e] = f;
  }
  __syncthreads();
  if (t < 32) {  // layer_norm over head_dim for q rows (0-15) and k rows (16-31)
    float* row = (t < 16) ? Q[t] : Kt[t - 16];
    const float* wv = (t < 16) ? qw : kw;
    const float* bv = (t < 16) ? qb : kb;
    float s = 0.0f, q2 = 0.0f;
    for (int e = 0; e < 64; ++e) { float x = row[e]; s += x; q2 += x * x; }
    float mu  = s * (1.0f / 64.0f);
    float var = q2 * (1.0f / 64.0f) - mu * mu;
    float rs  = rsqrtf(var + 1e-5f);
    for (int e = 0; e < 64; ++e) row[e] = (row[e] - mu) * rs * wv[e] + bv[e];
  }
  __syncthreads();
  {  // scores: thread (n, m)
    int n = t >> 4, m = t & 15;
    float d = 0.0f;
    for (int e = 0; e < 64; ++e) d += Q[n][e] * Kt[m][e];
    S_[n][m] = d * 0.125f;  // HD^-0.5
  }
  __syncthreads();
  if (t < 16) {  // softmax row t + per-head high-freq rescale
    float mx = -1e30f;
    for (int m = 0; m < 16; ++m) mx = fmaxf(mx, S_[t][m]);
    float sum = 0.0f;
    for (int m = 0; m < 16; ++m) { float e = __expf(S_[t][m] - mx); S_[t][m] = e; sum += e; }
    float inv = 1.0f / sum;
    float sc  = scale[head];
    const float low = 1.0f / 16.0f;
    for (int m = 0; m < 16; ++m) S_[t][m] = low + (S_[t][m] * inv - low) * sc;
  }
  __syncthreads();
#pragma unroll
  for (int j = 0; j < 4; ++j) {  // out = attn @ V
    int idx = t + j * 256;
    int n = idx >> 6, e = idx & 63;
    float acc = 0.0f;
    for (int m = 0; m < 16; ++m) acc += S_[n][m] * V[m][e];
    size_t g = ((size_t)(b * NTOK_ + n) * EMB_ + head * HD_ + e) * HWSZ_ + hw;
    outb[g] = f2bf(acc);
  }
}

// ---------------------------------------------------------------------------
extern "C" void kernel_launch(void* const* d_in, const int* in_sizes, int n_in,
                              void* d_out, int out_size, void* d_ws, size_t ws_size,
                              hipStream_t stream) {
  const float* x      = (const float*)d_in[0];
  const float* n1w    = (const float*)d_in[1];
  const float* n1b    = (const float*)d_in[2];
  const float* wqkv   = (const float*)d_in[3];
  const float* bqkv   = (const float*)d_in[4];
  const float* qnw    = (const float*)d_in[5];
  const float* qnb    = (const float*)d_in[6];
  const float* knw    = (const float*)d_in[7];
  const float* knb    = (const float*)d_in[8];
  const float* ascale = (const float*)d_in[9];
  const float* n2w    = (const float*)d_in[10];
  const float* n2b    = (const float*)d_in[11];
  const float* wout   = (const float*)d_in[12];
  const float* bout   = (const float*)d_in[13];
  (void)in_sizes; (void)n_in; (void)out_size; (void)ws_size;

  char* ws = (char*)d_ws;
  u16* xn      = (u16*)(ws + 0);           // 32*768*1024 bf16
  u16* wqkv_b  = (u16*)(ws + 50331648);    // 2304*768 bf16
  u16* wout_b  = (u16*)(ws + 53870592);    // 768*768 bf16
  u16* qkv_b   = (u16*)(ws + 55050240);    // 32*2304*1024 bf16
  u16* attnout = xn;                       // reuse xn region after GEMM1

  cvt_f32_bf16<<<(2304 * 768 + 255) / 256, 256, 0, stream>>>(wqkv, wqkv_b, 2304 * 768);
  cvt_f32_bf16<<<(768 * 768 + 255) / 256, 256, 0, stream>>>(wout, wout_b, 768 * 768);

  instnorm_f32<<<SAMP_ * EMB_, 256, 0, stream>>>(x, n1w, n1b, xn);

  gemm_bf16_wmma<true><<<dim3(SAMP_, 2304 / 128, HWSZ_ / 128), 256, 0, stream>>>(
      wqkv_b, xn, bqkv, (void*)qkv_b, 2304);

  attention_kernel<<<dim3(HWSZ_, HEADS_, BATCH_), 256, 0, stream>>>(
      qkv_b, attnout, qnw, qnb, knw, knb, ascale);

  instnorm_bf16<<<SAMP_ * EMB_, 256, 0, stream>>>(attnout, n2w, n2b);

  gemm_bf16_wmma<false><<<dim3(SAMP_, 768 / 128, HWSZ_ / 128), 256, 0, stream>>>(
      wout_b, attnout, bout, d_out, 768);
}
